// TransformerEncoder_40999757808050
// MI455X (gfx1250) — matrix operand order repaired and checked by reading, not compiled, not run on hardware
//
#include <hip/hip_runtime.h>
#include <math.h>

#define BB 32
#define NTOK 256
#define DD 240
#define HH 8
#define HDIM 30
#define DFFN 1024
#define NLAYERS 4
#define SEQ 1026
#define SEQP 1040
#define MROWS (BB*SEQP)       // 33280 rows, 2080 M-tiles, 520 M-chunks of 64
#define QTILES (SEQP/16)      // 65

typedef unsigned short u16;
typedef __attribute__((ext_vector_type(16))) __bf16 v16bf;
typedef __attribute__((ext_vector_type(8)))  float  v8f;
typedef __attribute__((ext_vector_type(8)))  u16    v8us;

union Frag { u16 u[16]; v8us h[2]; v16bf bf; };

__device__ __forceinline__ float bf2f(u16 v) { return __uint_as_float(((unsigned)v) << 16); }
__device__ __forceinline__ u16 f2bf(float f) {
    unsigned u = __float_as_uint(f);
    return (u16)((u + 0x7fffu + ((u >> 16) & 1u)) >> 16);
}
__device__ __forceinline__ v8us zero8() { v8us z = {0,0,0,0,0,0,0,0}; return z; }

// Max-reduce over each 16-lane DPP row entirely on the VALU pipe (no LDS).
// quad_perm(1,0,3,2)=xor1, quad_perm(2,3,0,1)=xor2, row_half_mirror=xor7,
// row_mirror=xor15 -- a complete 16-lane butterfly.
__device__ __forceinline__ float dpp_max16(float x) {
    int t;
    t = __builtin_amdgcn_update_dpp(__float_as_int(x), __float_as_int(x), 0xB1, 0xf, 0xf, false);
    x = fmaxf(x, __int_as_float(t));
    t = __builtin_amdgcn_update_dpp(__float_as_int(x), __float_as_int(x), 0x4E, 0xf, 0xf, false);
    x = fmaxf(x, __int_as_float(t));
    t = __builtin_amdgcn_update_dpp(__float_as_int(x), __float_as_int(x), 0x141, 0xf, 0xf, false);
    x = fmaxf(x, __int_as_float(t));
    t = __builtin_amdgcn_update_dpp(__float_as_int(x), __float_as_int(x), 0x140, 0xf, 0xf, false);
    x = fmaxf(x, __int_as_float(t));
    return x;
}

// A fragment: 16x32 bf16, row-major source. lanes0-15: K k0..k0+7 & k0+16..k0+23;
// lanes16-31: K k0+8..k0+15 & k0+24..k0+31. Chunks fully in [0,K) or zero (K%8==0).
__device__ __forceinline__ v16bf load_fragA(const u16* __restrict__ base, int ld,
                                            int row, int k0, int K, int hi) {
    const u16* rp = base + (size_t)row * ld;
    int ka = k0 + hi * 8;
    int kb = k0 + 16 + hi * 8;
    Frag f;
    f.h[0] = (ka + 8 <= K) ? *(const v8us*)(rp + ka) : zero8();
    f.h[1] = (kb + 8 <= K) ? *(const v8us*)(rp + kb) : zero8();
    return f.bf;
}

// B fragment: 32x16 bf16 from W^T (N x K row-major). lane col n holds
// K k0+hi*8 .. k0+hi*8+7 & k0+16+hi*8 .. k0+16+hi*8+7 in W^T row n.
__device__ __forceinline__ v16bf load_fragB(const u16* __restrict__ baseT, int ld,
                                            int nrow, int k0, int K, int hi) {
    const u16* rp = baseT + (size_t)nrow * ld;
    int ka = k0 + hi * 8;
    Frag f;
    f.h[0] = (ka + 8  <= K) ? *(const v8us*)(rp + ka)     : zero8();
    f.h[1] = (ka + 24 <= K) ? *(const v8us*)(rp + ka + 16) : zero8();
    return f.bf;
}

// ---------------- GEMM: C(MxN) = A(MxK,bf16) * W(KxN) + bias ----------------
// W given transposed bf16 (NxK). Wave computes 64x16 strip (4 M-tiles x 1 N-tile).
// EPI: 0 -> f32 out; 1 -> bf16 gelu(out); 2 -> bf16 out
template<int EPI>
__global__ __launch_bounds__(256) void gemm_kernel(
        const u16* __restrict__ A, int lda,
        const u16* __restrict__ Bt, int ldb,
        const float* __restrict__ bias,
        float* __restrict__ outF, u16* __restrict__ outH, int ldo,
        int Mchunks, int Ntiles, int K) {
    int wid = blockIdx.x * 8 + (threadIdx.x >> 5);
    if (wid >= Mchunks * Ntiles) return;
    int lane = threadIdx.x & 31;
    int l16 = lane & 15, hi = lane >> 4;
    int mc = wid / Ntiles, nt = wid % Ntiles;
    int m0 = mc * 64, n0 = nt * 16;

    v8f acc[4];
#pragma unroll
    for (int t = 0; t < 4; ++t)
#pragma unroll
        for (int r = 0; r < 8; ++r) acc[t][r] = 0.0f;

    for (int k0 = 0; k0 < K; k0 += 32) {
        if (k0 + 32 < K) {  // L2-resident prefetch of next K chunk
            __builtin_prefetch(Bt + (size_t)(n0 + l16) * ldb + k0 + 32, 0, 0);
            __builtin_prefetch(A + (size_t)(m0 + l16) * lda + k0 + 32, 0, 0);
        }
        v16bf bf_ = load_fragB(Bt, ldb, n0 + l16, k0, K, hi);
#pragma unroll
        for (int t = 0; t < 4; ++t) {
            v16bf af = load_fragA(A, lda, m0 + t * 16 + l16, k0, K, hi);
            acc[t] = __builtin_amdgcn_wmma_f32_16x16x32_bf16(
                false, af, false, bf_, (short)0, acc[t], false, false);
        }
    }

    float bcol = bias[n0 + l16];
#pragma unroll
    for (int t = 0; t < 4; ++t) {
#pragma unroll
        for (int r = 0; r < 8; ++r) {
            size_t row = (size_t)(m0 + t * 16 + r + hi * 8);
            size_t o = row * (size_t)ldo + (n0 + l16);
            float v = acc[t][r] + bcol;
            if (EPI == 0) outF[o] = v;
            else if (EPI == 1) outH[o] = f2bf(0.5f * v * (1.0f + erff(v * 0.70710678f)));
            else outH[o] = f2bf(v);
        }
    }
}

// ---------------- Flash attention: one wave per (b,h,query-tile) ----------------
// Q fragment carries exact zeros at K=30,31 (and the 1/sqrt(30) scale folded in at
// rotary time), so the K-side fragment loads unconditionally. P fragment carries
// zeros at keys 16..31, nullifying the V lanes 16..31. The V fragment's dead dim-30
// column is filled with bf16 1.0 so the o1 accumulator's column 30 maintains the
// online-softmax denominator for free on the matrix pipe (alpha rescale included).
// Row-max reduction runs on the VALU via DPP16 (no LDS traffic).
template<bool MASKED>
__device__ __forceinline__ void attn_tile(
        int kbase, size_t rowbase, int h, int l16, int hi,
        const Frag& qf, const u16* __restrict__ Kb, const u16* __restrict__ Vb,
        u16* pl, v8f& o0, v8f& o1, float* mstate) {
    // K^T fragment (B layout 32x16: K=dims, N=keys): two contiguous 16B loads
    Frag kf;
    {
        const u16* kr = Kb + (rowbase + kbase + l16) * DD + h * HDIM + hi * 8;
        kf.h[0] = *(const v8us*)(kr);
        kf.h[1] = *(const v8us*)(kr + 16);
    }
    v8f s;
#pragma unroll
    for (int r = 0; r < 8; ++r) s[r] = 0.f;
    s = __builtin_amdgcn_wmma_f32_16x16x32_bf16(false, qf.bf, false, kf.bf, (short)0, s, false, false);

    int key = kbase + l16;
#pragma unroll
    for (int r = 0; r < 8; ++r) {
        float sv = s[r];
        if (MASKED) { if (key >= SEQ) sv = -1.0e30f; }
        float tm = dpp_max16(sv);
        float nm = fmaxf(mstate[r], tm);
        float p = __expf(sv - nm);
        float alpha = __expf(mstate[r] - nm);
        mstate[r] = nm;
        o0[r] *= alpha; o1[r] *= alpha;
        pl[(r + hi * 8) * 16 + l16] = f2bf(p);
    }
    asm volatile("s_wait_dscnt 0" ::: "memory");

    // P fragment (A layout 16x32: rows=queries, K=keys 0..15, 16..31 zero)
    Frag pf;
#pragma unroll
    for (int j = 0; j < 16; ++j) pf.u[j] = hi ? (u16)0 : pl[l16 * 16 + j];

    // V fragments (B layout 32x16: K=keys, N=dims). Branchless clamped loads.
    // dim 30 (l16==14 of tile1) := 1.0 so o1 col30 accumulates sum(p).
    Frag vf0, vf1;
    int dd1 = (l16 < 14) ? 16 + l16 : 13;  // clamp to stay inside the head
    bool kill1 = hi || (l16 >= 14);
    u16 fill1 = (!hi && l16 == 14) ? (u16)0x3F80 : (u16)0;
#pragma unroll
    for (int j = 0; j < 16; ++j) {
        const u16* vr = Vb + (rowbase + kbase + j) * DD + h * HDIM;
        u16 a = vr[l16];
        u16 b_ = vr[dd1];
        vf0.u[j] = hi ? (u16)0 : a;
        vf1.u[j] = kill1 ? fill1 : b_;
    }
    o0 = __builtin_amdgcn_wmma_f32_16x16x32_bf16(false, pf.bf, false, vf0.bf, (short)0, o0, false, false);
    o1 = __builtin_amdgcn_wmma_f32_16x16x32_bf16(false, pf.bf, false, vf1.bf, (short)0, o1, false, false);
}

__global__ __launch_bounds__(256) void attention_kernel(
        const u16* __restrict__ Qb, const u16* __restrict__ Kb,
        const u16* __restrict__ Vb, u16* __restrict__ ATb) {
    __shared__ u16 pbuf[8 * 256];
    int w = threadIdx.x >> 5, lane = threadIdx.x & 31;
    int l16 = lane & 15, hi = lane >> 4;
    int wid = blockIdx.x * 8 + w;
    if (wid >= BB * HH * QTILES) return;
    int qt = wid % QTILES;
    int bh = wid / QTILES;
    int h = bh % HH, b = bh / HH;
    size_t rowbase = (size_t)b * SEQP;
    int q0 = qt * 16;
    u16* pl = pbuf + w * 256;

    // Q fragment (A layout, 16x32): unconditional loads, then zero padded dims 30,31
    Frag qf;
    {
        const u16* qr = Qb + (rowbase + q0 + l16) * DD + h * HDIM;
        qf.h[0] = *(const v8us*)(qr + hi * 8);
        qf.h[1] = *(const v8us*)(qr + 16 + hi * 8);
        if (hi) { qf.u[14] = 0; qf.u[15] = 0; }   // K=30,31 -> 0
    }

    v8f o0, o1;
    float mstate[8];
#pragma unroll
    for (int r = 0; r < 8; ++r) { o0[r] = 0.f; o1[r] = 0.f; mstate[r] = -3.0e38f; }

    for (int kt = 0; kt < QTILES - 1; ++kt)
        attn_tile<false>(kt * 16, rowbase, h, l16, hi, qf, Kb, Vb, pl, o0, o1, mstate);
    attn_tile<true>((QTILES - 1) * 16, rowbase, h, l16, hi, qf, Kb, Vb, pl, o0, o1, mstate);

#pragma unroll
    for (int r = 0; r < 8; ++r) {
        size_t row = rowbase + q0 + r + hi * 8;
        float l = __shfl(o1[r], 14, 16);   // softmax denominator from V's ones column
        float inv = 1.0f / l;
        ATb[row * DD + h * HDIM + l16] = f2bf(o0[r] * inv);
        if (l16 < HDIM - 16) ATb[row * DD + h * HDIM + 16 + l16] = f2bf(o1[r] * inv);
    }
}

// ---------------- residual add + layernorm (one wave per row) ----------------
__global__ __launch_bounds__(256) void add_ln_kernel(
        const float* __restrict__ X, const float* __restrict__ Y,
        const float* __restrict__ g, const float* __restrict__ be,
        float* __restrict__ outF, u16* __restrict__ outH) {
    int row = blockIdx.x * 8 + (threadIdx.x >> 5);
    if (row >= MROWS) return;
    int lane = threadIdx.x & 31;
    const float* xr = X + (size_t)row * DD;
    const float* yr = Y + (size_t)row * DD;
    float v[8]; float sum = 0.f, sq = 0.f;
#pragma unroll
    for (int j = 0; j < 8; ++j) {
        int c = lane * 8 + j;
        float t = (c < DD) ? xr[c] + yr[c] : 0.f;
        v[j] = t; sum += t; sq += t * t;
    }
    for (int off = 1; off < 32; off <<= 1) { sum += __shfl_xor(sum, off, 32); sq += __shfl_xor(sq, off, 32); }
    float mu = sum * (1.0f / DD);
    float var = sq * (1.0f / DD) - mu * mu;
    float inv = rsqrtf(var + 1e-5f);
#pragma unroll
    for (int j = 0; j < 8; ++j) {
        int c = lane * 8 + j;
        if (c < DD) {
            float o = (v[j] - mu) * inv * g[c] + be[c];
            outF[(size_t)row * DD + c] = o;
            outH[(size_t)row * DD + c] = f2bf(o);
        }
    }
}

// ---------------- rotary 3D PE on Q,K (bf16 in/out) ----------------
// Q additionally gets the 1/sqrt(HDIM) attention scale folded in.
__global__ void rotary_kernel(const u16* __restrict__ Qr, const u16* __restrict__ Kr,
                              const float* __restrict__ coords,
                              u16* __restrict__ Qo, u16* __restrict__ Ko) {
    const float invf[5] = {1.0f, 0.15848932f, 0.025118863f, 0.0039810716f, 0.00063095737f};
    const float scl = 0.18257419f;  // 1/sqrt(30)
    size_t total = (size_t)MROWS * DD;
    for (size_t idx = (size_t)blockIdx.x * blockDim.x + threadIdx.x; idx < total;
         idx += (size_t)gridDim.x * blockDim.x) {
        int c = (int)(idx % DD);
        size_t row = idx / DD;
        int h = c / HDIM, hc = c % HDIM;
        int a = hc / 10, r = hc % 10;
        int j = (r < 5) ? r : r - 5;
        float th = coords[row * 3 + a] * invf[j];
        float sn, cs; __sincosf(th, &sn, &cs);
        size_t base = row * DD + h * HDIM + a * 10;
        float qx1 = bf2f(Qr[base + j]), qx2 = bf2f(Qr[base + j + 5]);
        float kx1 = bf2f(Kr[base + j]), kx2 = bf2f(Kr[base + j + 5]);
        float qo = (r < 5) ? qx1 * cs - qx2 * sn : qx1 * sn + qx2 * cs;
        float ko = (r < 5) ? kx1 * cs - kx2 * sn : kx1 * sn + kx2 * cs;
        Qo[idx] = f2bf(qo * scl);
        Ko[idx] = f2bf(ko);
    }
}

// ---------------- assemble src + coords ----------------
__global__ void assemble_kernel(const float* __restrict__ hand_t, const float* __restrict__ head_t,
                                const float* __restrict__ hand_m1, const float* __restrict__ head_m1,
                                const float* __restrict__ st, const float* __restrict__ sm1,
                                float* __restrict__ X, u16* __restrict__ Xb) {
    size_t total = (size_t)MROWS * DD;
    for (size_t idx = (size_t)blockIdx.x * blockDim.x + threadIdx.x; idx < total;
         idx += (size_t)gridDim.x * blockDim.x) {
        int c = (int)(idx % DD);
        size_t row = idx / DD;
        int b = (int)(row / SEQP), s = (int)(row % SEQP);
        float v = 0.f;
        if (s == 0)            v = sm1[(size_t)b * DD + c];
        else if (s <= 256)     v = hand_m1[((size_t)b * NTOK + (s - 1)) * DD + c];
        else if (s <= 512)     v = head_m1[((size_t)b * NTOK + (s - 257)) * DD + c];
        else if (s == 513)     v = st[(size_t)b * DD + c];
        else if (s <= 769)     v = hand_t[((size_t)b * NTOK + (s - 514)) * DD + c];
        else if (s < SEQ)      v = head_t[((size_t)b * NTOK + (s - 770)) * DD + c];
        X[idx] = v;
        Xb[idx] = f2bf(v);
    }
}

__global__ void coords_kernel(const float* __restrict__ c_hand_t, const float* __restrict__ c_head_t,
                              const float* __restrict__ c_hand_m1, const float* __restrict__ c_head_m1,
                              float* __restrict__ C) {
    size_t total = (size_t)MROWS * 3;
    for (size_t idx = (size_t)blockIdx.x * blockDim.x + threadIdx.x; idx < total;
         idx += (size_t)gridDim.x * blockDim.x) {
        int a = (int)(idx % 3);
        size_t row = idx / 3;
        int b = (int)(row / SEQP), s = (int)(row % SEQP);
        float v = 0.f;
        if (s >= 1 && s <= 256)        v = c_hand_m1[((size_t)b * NTOK + (s - 1)) * 3 + a];
        else if (s >= 257 && s <= 512) v = c_head_m1[((size_t)b * NTOK + (s - 257)) * 3 + a];
        else if (s >= 514 && s <= 769) v = c_hand_t[((size_t)b * NTOK + (s - 514)) * 3 + a];
        else if (s >= 770 && s < SEQ)  v = c_head_t[((size_t)b * NTOK + (s - 770)) * 3 + a];
        C[idx] = v;
    }
}

// ---------------- weight transpose + bf16 convert: W(KxN) -> Wt(NxK) ----------------
__global__ void transpose_bf16_kernel(const float* __restrict__ W, u16* __restrict__ Wt, int K, int N) {
    size_t total = (size_t)K * N;
    for (size_t idx = (size_t)blockIdx.x * blockDim.x + threadIdx.x; idx < total;
         idx += (size_t)gridDim.x * blockDim.x) {
        int n = (int)(idx / K), k = (int)(idx % K);
        Wt[idx] = f2bf(W[(size_t)k * N + n]);
    }
}

__global__ void output_kernel(const float* __restrict__ X, float* __restrict__ out) {
    size_t total = (size_t)BB * 513 * DD;
    for (size_t idx = (size_t)blockIdx.x * blockDim.x + threadIdx.x; idx < total;
         idx += (size_t)gridDim.x * blockDim.x) {
        int c = (int)(idx % DD);
        size_t rt = idx / DD;
        int t = (int)(rt % 513);
        int b = (int)(rt / 513);
        out[idx] = X[((size_t)b * SEQP + 513 + t) * DD + c];
    }
}

// ---------------- host ----------------
static inline size_t align256(size_t x) { return (x + 255) & ~(size_t)255; }

extern "C" void kernel_launch(void* const* d_in, const int* in_sizes, int n_in,
                              void* d_out, int out_size, void* d_ws, size_t ws_size,
                              hipStream_t stream) {
    (void)in_sizes; (void)n_in; (void)out_size; (void)ws_size;
    const float* hand_t  = (const float*)d_in[0];
    const float* head_t  = (const float*)d_in[1];
    const float* hand_m1 = (const float*)d_in[2];
    const float* head_m1 = (const float*)d_in[3];
    const float* c_hand_t  = (const float*)d_in[4];
    const float* c_head_t  = (const float*)d_in[5];
    const float* c_hand_m1 = (const float*)d_in[6];
    const float* c_head_m1 = (const float*)d_in[7];
    const float* state_t  = (const float*)d_in[8];
    const float* state_m1 = (const float*)d_in[9];
    const float* Wq = (const float*)d_in[10];
    const float* Wk = (const float*)d_in[11];
    const float* Wv = (const float*)d_in[12];
    const float* Wo = (const float*)d_in[13];
    const float* W1 = (const float*)d_in[14];
    const float* W2 = (const float*)d_in[15];
    const float* bq = (const float*)d_in[16];
    const float* bk = (const float*)d_in[17];
    const float* bv = (const float*)d_in[18];
    const float* bo = (const float*)d_in[19];
    const float* b1 = (const float*)d_in[20];
    const float* b2 = (const float*)d_in[21];
    const float* g1 = (const float*)d_in[22];
    const float* be1 = (const float*)d_in[23];
    const float* g2 = (const float*)d_in[24];
    const float* be2 = (const float*)d_in[25];

    char* ws = (char*)d_ws;
    size_t off = 0;
    auto alloc = [&](size_t bytes) -> char* { char* p = ws + off; off += align256(bytes); return p; };
    const size_t NE = (size_t)MROWS * DD;

    float* X32   = (float*)alloc(NE * 4);
    float* X2_32 = (float*)alloc(NE * 4);
    float* AD    = (float*)alloc(NE * 4);          // O-proj out, reused for FFN2 out
    u16* Xb   = (u16*)alloc(NE * 2);
    u16* X2b  = (u16*)alloc(NE * 2);
    u16* Qraw = (u16*)alloc(NE * 2);
    u16* Kraw = (u16*)alloc(NE * 2);
    u16* Vb   = (u16*)alloc(NE * 2);
    u16* Qb   = (u16*)alloc(NE * 2);
    u16* Kb   = (u16*)alloc(NE * 2);
    u16* ATb  = (u16*)alloc(NE * 2);
    u16* Hb   = (u16*)alloc((size_t)MROWS * DFFN * 2);
    float* Cds = (float*)alloc((size_t)MROWS * 3 * 4);
    u16* WqT = (u16*)alloc((size_t)NLAYERS * DD * DD * 2);
    u16* WkT = (u16*)alloc((size_t)NLAYERS * DD * DD * 2);
    u16* WvT = (u16*)alloc((size_t)NLAYERS * DD * DD * 2);
    u16* WoT = (u16*)alloc((size_t)NLAYERS * DD * DD * 2);
    u16* W1T = (u16*)alloc((size_t)NLAYERS * DFFN * DD * 2);  // (DFF x D)
    u16* W2T = (u16*)alloc((size_t)NLAYERS * DD * DFFN * 2);  // (D x DFF)

    const int TB = 256;
    for (int l = 0; l < NLAYERS; ++l) {
        size_t wdd = (size_t)l * DD * DD;
        transpose_bf16_kernel<<<225, TB, 0, stream>>>(Wq + wdd, WqT + wdd, DD, DD);
        transpose_bf16_kernel<<<225, TB, 0, stream>>>(Wk + wdd, WkT + wdd, DD, DD);
        transpose_bf16_kernel<<<225, TB, 0, stream>>>(Wv + wdd, WvT + wdd, DD, DD);
        transpose_bf16_kernel<<<225, TB, 0, stream>>>(Wo + wdd, WoT + wdd, DD, DD);
        transpose_bf16_kernel<<<960, TB, 0, stream>>>(W1 + (size_t)l * DD * DFFN, W1T + (size_t)l * DFFN * DD, DD, DFFN);
        transpose_bf16_kernel<<<960, TB, 0, stream>>>(W2 + (size_t)l * DFFN * DD, W2T + (size_t)l * DD * DFFN, DFFN, DD);
    }
    assemble_kernel<<<4096, TB, 0, stream>>>(hand_t, head_t, hand_m1, head_m1, state_t, state_m1, X32, Xb);
    coords_kernel<<<512, TB, 0, stream>>>(c_hand_t, c_head_t, c_hand_m1, c_head_m1, Cds);

    const int Mchunks = MROWS / 64;  // 520
    auto gemm = [&](int epi, const u16* A, int lda, const u16* Bt, int ldb,
                    const float* bias, float* oF, u16* oH, int N, int K) {
        int Ntiles = N / 16;
        int blocks = (Mchunks * Ntiles + 7) / 8;
        if (epi == 0)      gemm_kernel<0><<<blocks, TB, 0, stream>>>(A, lda, Bt, ldb, bias, oF, oH, N, Mchunks, Ntiles, K);
        else if (epi == 1) gemm_kernel<1><<<blocks, TB, 0, stream>>>(A, lda, Bt, ldb, bias, oF, oH, N, Mchunks, Ntiles, K);
        else               gemm_kernel<2><<<blocks, TB, 0, stream>>>(A, lda, Bt, ldb, bias, oF, oH, N, Mchunks, Ntiles, K);
    };

    for (int l = 0; l < NLAYERS; ++l) {
        size_t wdd = (size_t)l * DD * DD;
        gemm(2, Xb, DD, WqT + wdd, DD, bq + (size_t)l * DD, nullptr, Qraw, DD, DD);
        gemm(2, Xb, DD, WkT + wdd, DD, bk + (size_t)l * DD, nullptr, Kraw, DD, DD);
        gemm(2, Xb, DD, WvT + wdd, DD, bv + (size_t)l * DD, nullptr, Vb,   DD, DD);
        rotary_kernel<<<4096, TB, 0, stream>>>(Qraw, Kraw, Cds, Qb, Kb);
        attention_kernel<<<(BB * HH * QTILES) / 8, TB, 0, stream>>>(Qb, Kb, Vb, ATb);
        gemm(0, ATb, DD, WoT + wdd, DD, bo + (size_t)l * DD, AD, nullptr, DD, DD);
        add_ln_kernel<<<MROWS / 8, TB, 0, stream>>>(X32, AD, g1 + (size_t)l * DD, be1 + (size_t)l * DD, X2_32, X2b);
        gemm(1, X2b, DD, W1T + (size_t)l * DFFN * DD, DD, b1 + (size_t)l * DFFN, nullptr, Hb, DFFN, DD);
        gemm(0, Hb, DFFN, W2T + (size_t)l * DD * DFFN, DFFN, b2 + (size_t)l * DD, AD, nullptr, DD, DFFN);
        add_ln_kernel<<<MROWS / 8, TB, 0, stream>>>(X2_32, AD, g2 + (size_t)l * DD, be2 + (size_t)l * DD, X32, Xb);
    }
    output_kernel<<<4096, TB, 0, stream>>>(X32, (float*)d_out);
}
